// Pseudo_MultiHeadSelfAttention_21328807592206
// MI455X (gfx1250) — compile-verified, matrix operand order checked by reading, not run on hardware
//
#include <hip/hip_runtime.h>

typedef __attribute__((ext_vector_type(8)))  __bf16 v8bf;
typedef __attribute__((ext_vector_type(16))) __bf16 v16bf;
typedef __attribute__((ext_vector_type(8)))  float  v8f;

#define BATCH 2
#define SEQ   2048
#define EMB   1024
#define NHEAD 16
#define HDIM  64

// ---------------- device helpers ----------------

__device__ __forceinline__ __bf16 f2bf(float f) {
  // round-to-nearest-even f32 -> bf16
  unsigned u = __float_as_uint(f);
  unsigned r = u + 0x7FFFu + ((u >> 16) & 1u);
  unsigned short s = (unsigned short)(r >> 16);
  __bf16 b; __builtin_memcpy(&b, &s, 2); return b;
}

// Async global->LDS copy of 16 bytes (CDNA5, tracked by ASYNCcnt).
// VDST VGPR holds the wave-relative LDS byte address; for HIP generic shared
// pointers the low 32 address bits ARE the LDS address (flat aperture rule).
__device__ __forceinline__ void async_copy_b128(const void* g, void* l) {
  unsigned lds_addr = (unsigned)(size_t)l;
  asm volatile("global_load_async_to_lds_b128 %0, %1, off"
               :: "v"(lds_addr), "v"(g) : "memory");
}

__device__ __forceinline__ void wait_async0() {
  asm volatile("s_wait_asynccnt 0x0" ::: "memory");
}

// Cooperatively stage an R x C bf16 tile (global row stride ld elems) into LDS
// (contiguous rows of C elems). 128 threads, 16B per async op.
// R*C*2 must be a multiple of 128*16 so the loop has a static trip count
// (no divergence, no EXEC manipulation around the async ops).
template <int R, int C>
__device__ __forceinline__ void stage_tile(const __bf16* g, int ld, __bf16* l) {
  constexpr int CPR = (C * 2) / 16;          // 16B chunks per row
  constexpr int ITERS = (R * CPR) / 128;
  static_assert(R * CPR == ITERS * 128, "tile must fill all 128 threads evenly");
#pragma unroll
  for (int t = 0; t < ITERS; ++t) {
    int i = (int)threadIdx.x + t * 128;
    int row = i / CPR, ch = i % CPR;
    async_copy_b128(g + (size_t)row * ld + ch * 8, l + row * C + ch * 8);
  }
}

// Load a 16x32 bf16 tile (row-major, row stride ld) into WMMA A/B operand layout.
// ISA layout (16-bit, wave32): lanes 0-15 = rows 0-15 holding K {0..7,16..23},
// lanes 16-31 = rows 0-15 holding K {8..15,24..31}. Two 16B loads per lane.
__device__ __forceinline__ v16bf load_frag(const __bf16* tile, int ld) {
  const int lane = threadIdx.x & 31;
  const int r = lane & 15;
  const int hh = lane >> 4;
  const __bf16* p = tile + r * ld + hh * 8;
  v8bf lo = *(const v8bf*)(p);
  v8bf hi = *(const v8bf*)(p + 16);
  v16bf o;
#pragma unroll
  for (int i = 0; i < 8; ++i) { o[i] = lo[i]; o[i + 8] = hi[i]; }
  return o;
}

__device__ __forceinline__ v8f wmma_bf16(v16bf a, v16bf b, v8f c) {
  return __builtin_amdgcn_wmma_f32_16x16x32_bf16(false, a, false, b, (short)0, c,
                                                 false, false);
}

// C layout: VGPR v -> row (v + 8*(lane>=16)), col = lane&15
__device__ __forceinline__ void store_acc_bf16(const v8f& c, __bf16* p, int rs, int cs) {
  const int lane = threadIdx.x & 31;
  const int col = lane & 15;
  const int half = lane >> 4;
#pragma unroll
  for (int v = 0; v < 8; ++v)
    p[(size_t)(v + 8 * half) * rs + (size_t)col * cs] = f2bf(c[v]);
}

// One pipelined GEMM K-step: wait for current tiles, prefetch next into the
// statically-named other buffer, run 4 WMMAs from LDS.
__device__ __forceinline__ void gemm_step(const __bf16* Ag, const __bf16* Bg,
                                          int lda, int ldb, int knext, int K,
                                          const __bf16* curA, const __bf16* curB,
                                          __bf16* nxtA, __bf16* nxtB, v8f acc[4]) {
  const int wave = threadIdx.x >> 5;
  wait_async0();
  __syncthreads();
  if (knext < K) {
    stage_tile<64, 32>(Ag + knext, lda, nxtA);
    stage_tile<64, 32>(Bg + knext, ldb, nxtB);
  }
  v16bf a = load_frag(curA + wave * 16 * 32, 32);
#pragma unroll
  for (int j = 0; j < 4; ++j) {
    v16bf b = load_frag(curB + j * 16 * 32, 32);
    acc[j] = wmma_bf16(a, b, acc[j]);
  }
}

// Double-buffered LDS-staged 64x64 GEMM tile: acc[4] += A(64xK) * BT(64xK)^T
// K must be a multiple of 64 (2-deep unroll with static buffer names).
__device__ __forceinline__ void gemm_tiles(const __bf16* Ag, const __bf16* Bg,
                                           int lda, int ldb, int K,
                                           __bf16 (*lA)[64 * 32],
                                           __bf16 (*lB)[64 * 32], v8f acc[4]) {
  stage_tile<64, 32>(Ag, lda, lA[0]);
  stage_tile<64, 32>(Bg, ldb, lB[0]);
  for (int k = 0; k < K; k += 64) {
    gemm_step(Ag, Bg, lda, ldb, k + 32, K, lA[0], lB[0], lA[1], lB[1], acc);
    gemm_step(Ag, Bg, lda, ldb, k + 64, K, lA[1], lB[1], lA[0], lB[0], acc);
  }
}

// ---------------- prep kernels ----------------

__global__ void k_cvt(const float* __restrict__ src, __bf16* __restrict__ dst, int n) {
  int i = blockIdx.x * blockDim.x + threadIdx.x;
  if (i < n) dst[i] = f2bf(src[i]);
}

// dst[c*rows + r] = src[r*cols + c]
__global__ void k_transpose_cvt(const float* __restrict__ src, __bf16* __restrict__ dst,
                                int rows, int cols) {
  int i = blockIdx.x * blockDim.x + threadIdx.x;
  if (i < rows * cols) {
    int r = i / cols, c = i % cols;
    dst[(size_t)c * rows + r] = f2bf(src[i]);
  }
}

// src [N][D][H] -> dst [N][H][D] (scaled)
__global__ void k_head_transpose(const float* __restrict__ src, __bf16* __restrict__ dst,
                                 float scale) {
  int i = blockIdx.x * blockDim.x + threadIdx.x;  // N*64*64
  int n = i >> 12, rem = i & 4095, hcol = rem >> 6, d = rem & 63;
  dst[i] = f2bf(src[(n << 12) + (d << 6) + hcol] * scale);
}

// ---------------- GEMM: C = A * BT^T (bf16 out) ----------------
__global__ void k_gemm_bf(const __bf16* __restrict__ A, const __bf16* __restrict__ BT,
                          __bf16* __restrict__ C, int lda, int ldb, int ldc, int K) {
  __shared__ __align__(16) __bf16 lA[2][64 * 32];
  __shared__ __align__(16) __bf16 lB[2][64 * 32];
  const int wave = threadIdx.x >> 5;
  const int m0 = blockIdx.x * 64 + wave * 16;
  const int n0 = blockIdx.y * 64;
  v8f acc[4] = {};
  gemm_tiles(A + (size_t)blockIdx.x * 64 * lda, BT + (size_t)n0 * ldb, lda, ldb, K,
             lA, lB, acc);
#pragma unroll
  for (int j = 0; j < 4; ++j)
    store_acc_bf16(acc[j], C + (size_t)m0 * ldc + n0 + j * 16, ldc, 1);
}

// ---------------- GEMM + bias epilogue (f32 out) ----------------
__global__ void k_out(const __bf16* __restrict__ A, const __bf16* __restrict__ BT,
                      const float* __restrict__ bias, float* __restrict__ C) {
  __shared__ __align__(16) __bf16 lA[2][64 * 32];
  __shared__ __align__(16) __bf16 lB[2][64 * 32];
  const int wave = threadIdx.x >> 5;
  const int lane = threadIdx.x & 31;
  const int col = lane & 15, half = lane >> 4;
  const int m0 = blockIdx.x * 64 + wave * 16;
  const int n0 = blockIdx.y * 64;
  v8f acc[4] = {};
  gemm_tiles(A + (size_t)blockIdx.x * 64 * EMB, BT + (size_t)n0 * EMB, EMB, EMB, EMB,
             lA, lB, acc);
#pragma unroll
  for (int j = 0; j < 4; ++j) {
    float bv = bias[n0 + j * 16 + col];
#pragma unroll
    for (int v = 0; v < 8; ++v)
      C[(size_t)(m0 + v + 8 * half) * EMB + n0 + j * 16 + col] = acc[j][v] + bv;
  }
}

// ---------------- fused q = h*Wa (scaled), vT = (h*Wv)^T per head ----------------
__global__ void k_qv(const __bf16* __restrict__ h, const __bf16* __restrict__ awT,
                     const __bf16* __restrict__ vwT, __bf16* __restrict__ q,
                     __bf16* __restrict__ vT) {
  const int bn = blockIdx.y, b = bn >> 4, n = bn & 15;
  const int wave = threadIdx.x >> 5;
  const int s0 = blockIdx.x * 64 + wave * 16;
  const __bf16* hb = h + (size_t)b * SEQ * EMB + n * HDIM;
  const __bf16* aw = awT + (size_t)n * HDIM * HDIM;
  const __bf16* vw = vwT + (size_t)n * HDIM * HDIM;
  v8f qa[4] = {}, va[4] = {};
#pragma unroll
  for (int kc = 0; kc < 2; ++kc) {
    v16bf a = load_frag(hb + (size_t)s0 * EMB + kc * 32, EMB);
#pragma unroll
    for (int j = 0; j < 4; ++j) {
      qa[j] = wmma_bf16(a, load_frag(aw + (j * 16) * HDIM + kc * 32, HDIM), qa[j]);
      va[j] = wmma_bf16(a, load_frag(vw + (j * 16) * HDIM + kc * 32, HDIM), va[j]);
    }
  }
  __bf16* qb = q + (size_t)b * SEQ * EMB + n * HDIM;
  __bf16* vb = vT + (size_t)bn * HDIM * SEQ;  // [d][s]
#pragma unroll
  for (int j = 0; j < 4; ++j) {
    store_acc_bf16(qa[j], qb + (size_t)s0 * EMB + j * 16, EMB, 1);
    store_acc_bf16(va[j], vb + (size_t)(j * 16) * SEQ + s0, 1, SEQ);  // transposed
  }
}

// ---------------- flash attention ----------------

struct FlashState {
  v16bf aq0, aq1;
  v8f o[4];
  float mrow[8], lrow[8];
};

// One 32-key flash block: wait for staged K/V, prefetch next block into the
// statically-named other buffers, score WMMAs, online softmax, P via LDS, O WMMAs.
__device__ __forceinline__ void flash_step(const __bf16* kb, const __bf16* vb,
                                           int rnext, const __bf16* lKc,
                                           const __bf16* lVc, __bf16* lKn,
                                           __bf16* lVn, __bf16* lp, FlashState& st) {
  const int lane = threadIdx.x & 31;
  const int col = lane & 15, half = lane >> 4;

  wait_async0();
  __syncthreads();
  if (rnext < SEQ) {
    stage_tile<32, 64>(kb + (size_t)rnext * EMB, EMB, lKn);
    stage_tile<64, 32>(vb + rnext, SEQ, lVn);
  }

  v8f sc0 = {}, sc1 = {};
  sc0 = wmma_bf16(st.aq0, load_frag(lKc, 64), sc0);
  sc0 = wmma_bf16(st.aq1, load_frag(lKc + 32, 64), sc0);
  sc1 = wmma_bf16(st.aq0, load_frag(lKc + 16 * 64, 64), sc1);
  sc1 = wmma_bf16(st.aq1, load_frag(lKc + 16 * 64 + 32, 64), sc1);

  float oscale[8];
#pragma unroll
  for (int v = 0; v < 8; ++v) {
    float cmax = fmaxf(sc0[v], sc1[v]);
#pragma unroll
    for (int off = 1; off < 16; off <<= 1)
      cmax = fmaxf(cmax, __shfl_xor(cmax, off, 32));
    float mnew = fmaxf(st.mrow[v], cmax);
    float p0 = __expf(sc0[v] - mnew);
    float p1 = __expf(sc1[v] - mnew);
    float rs = p0 + p1;
#pragma unroll
    for (int off = 1; off < 16; off <<= 1)
      rs += __shfl_xor(rs, off, 32);
    float s = __expf(st.mrow[v] - mnew);
    st.lrow[v] = st.lrow[v] * s + rs;
    st.mrow[v] = mnew;
    oscale[v] = s;
    sc0[v] = p0; sc1[v] = p1;
  }
#pragma unroll
  for (int j = 0; j < 4; ++j)
#pragma unroll
    for (int v = 0; v < 8; ++v) st.o[j][v] *= oscale[v];

  // P (C-layout) -> LDS -> A-operand layout (per-wave scratch, same-wave RAW)
#pragma unroll
  for (int v = 0; v < 8; ++v) {
    int m = v + 8 * half;
    lp[m * 32 + col]      = f2bf(sc0[v]);
    lp[m * 32 + 16 + col] = f2bf(sc1[v]);
  }
  asm volatile("s_wait_dscnt 0" ::: "memory");
  v16bf ap = load_frag(lp, 32);
#pragma unroll
  for (int j = 0; j < 4; ++j)
    st.o[j] = wmma_bf16(ap, load_frag(lVc + (j * 16) * 32, 32), st.o[j]);
}

// per (b, head, 64-query-row WG); K/V async-staged into LDS, double-buffered.
__global__ void k_attn(const __bf16* __restrict__ q, const __bf16* __restrict__ h,
                       const __bf16* __restrict__ vT, __bf16* __restrict__ attn) {
  __shared__ __align__(16) __bf16 lK[2][32 * 64];  // 32 keys x 64 dims
  __shared__ __align__(16) __bf16 lV[2][64 * 32];  // 64 dims x 32 keys (V^T slice)
  __shared__ __align__(32) __bf16 ldsP[4][16 * 32];
  const int bn = blockIdx.y, b = bn >> 4, n = bn & 15;
  const int wave = threadIdx.x >> 5;
  const int lane = threadIdx.x & 31;
  const int col = lane & 15, half = lane >> 4;
  const int s0 = blockIdx.x * 64 + wave * 16;
  const __bf16* qb = q + (size_t)b * SEQ * EMB + n * HDIM;
  const __bf16* kb = h + (size_t)b * SEQ * EMB + n * HDIM;
  const __bf16* vb = vT + (size_t)bn * HDIM * SEQ;

  FlashState st;
  st.aq0 = load_frag(qb + (size_t)s0 * EMB, EMB);
  st.aq1 = load_frag(qb + (size_t)s0 * EMB + 32, EMB);
#pragma unroll
  for (int j = 0; j < 4; ++j) st.o[j] = v8f{};
#pragma unroll
  for (int v = 0; v < 8; ++v) { st.mrow[v] = -3.0e38f; st.lrow[v] = 0.f; }

  stage_tile<32, 64>(kb, EMB, lK[0]);
  stage_tile<64, 32>(vb, SEQ, lV[0]);
  __bf16* lp = ldsP[wave];

  for (int r0 = 0; r0 < SEQ; r0 += 64) {
    flash_step(kb, vb, r0 + 32, lK[0], lV[0], lK[1], lV[1], lp, st);
    flash_step(kb, vb, r0 + 64, lK[1], lV[1], lK[0], lV[0], lp, st);
  }

  __bf16* ab = attn + (size_t)b * SEQ * EMB + n * HDIM;
#pragma unroll
  for (int j = 0; j < 4; ++j)
#pragma unroll
    for (int v = 0; v < 8; ++v) {
      int m = v + 8 * half;
      ab[(size_t)(s0 + m) * EMB + j * 16 + col] = f2bf(st.o[j][v] / st.lrow[v]);
    }
}

// ---------------- launcher ----------------

extern "C" void kernel_launch(void* const* d_in, const int* in_sizes, int n_in,
                              void* d_out, int out_size, void* d_ws, size_t ws_size,
                              hipStream_t stream) {
  (void)in_sizes; (void)n_in; (void)out_size; (void)ws_size;
  const float* x     = (const float*)d_in[0];
  const float* projw = (const float*)d_in[1];
  const float* attnw = (const float*)d_in[2];
  const float* valw  = (const float*)d_in[3];
  const float* outw  = (const float*)d_in[4];
  const float* outb  = (const float*)d_in[5];
  float* out = (float*)d_out;

  char* ws = (char*)d_ws;
  auto alloc = [&](size_t bytes) {
    char* p = ws;
    ws += (bytes + 255) & ~(size_t)255;
    return p;
  };
  const size_t BSE = (size_t)BATCH * SEQ * EMB;
  __bf16* xb  = (__bf16*)alloc(BSE * 2);
  __bf16* hb  = (__bf16*)alloc(BSE * 2);
  __bf16* qb  = (__bf16*)alloc(BSE * 2);
  __bf16* vT  = (__bf16*)alloc((size_t)BATCH * NHEAD * HDIM * SEQ * 2);
  __bf16* ab  = (__bf16*)alloc(BSE * 2);
  __bf16* wT  = (__bf16*)alloc((size_t)EMB * EMB * 2);
  __bf16* awT = (__bf16*)alloc((size_t)NHEAD * HDIM * HDIM * 2);
  __bf16* vwT = (__bf16*)alloc((size_t)NHEAD * HDIM * HDIM * 2);
  __bf16* owb = (__bf16*)alloc((size_t)EMB * EMB * 2);

  const int nx = BATCH * SEQ * EMB;
  k_cvt<<<nx / 256, 256, 0, stream>>>(x, xb, nx);
  k_cvt<<<EMB * EMB / 256, 256, 0, stream>>>(outw, owb, EMB * EMB);
  k_transpose_cvt<<<EMB * EMB / 256, 256, 0, stream>>>(projw, wT, EMB, EMB);
  k_head_transpose<<<NHEAD * HDIM * HDIM / 256, 256, 0, stream>>>(attnw, awT,
                                                                  1.0f / (float)HDIM);
  k_head_transpose<<<NHEAD * HDIM * HDIM / 256, 256, 0, stream>>>(valw, vwT, 1.0f);

  dim3 g1(BATCH * SEQ / 64, EMB / 64);
  k_gemm_bf<<<g1, 128, 0, stream>>>(xb, wT, hb, EMB, EMB, EMB, EMB);

  dim3 g2(SEQ / 64, BATCH * NHEAD);
  k_qv<<<g2, 128, 0, stream>>>(hb, awT, vwT, qb, vT);
  k_attn<<<g2, 128, 0, stream>>>(qb, hb, vT, ab);
  k_out<<<g1, 128, 0, stream>>>(ab, owb, outb, out);
}
